// PositionAttention_37804302139763
// MI455X (gfx1250) — matrix operand order repaired and checked by reading, not compiled, not run on hardware
//
#include <hip/hip_runtime.h>
#include <hip/hip_bf16.h>

#define B_   32
#define C_   512
#define P1_  1024
#define P2_  2048
#define CQK_ 64

typedef __attribute__((ext_vector_type(16))) _Float16 v16h;
typedef __attribute__((ext_vector_type(8)))  _Float16 v8h;
typedef __attribute__((ext_vector_type(8)))  float    v8f;
typedef __attribute__((ext_vector_type(4)))  float    v4f;

__device__ __forceinline__ v16h cat8(v8h a, v8h b) {
  return __builtin_shufflevector(a, b, 0, 1, 2, 3, 4, 5, 6, 7, 8, 9, 10, 11,
                                 12, 13, 14, 15);
}

// ---------------------------------------------------------------------------
// fp32 -> f16 weight conversion (tiny: <= 512x512)
// ---------------------------------------------------------------------------
__global__ __launch_bounds__(256) void pa_cvt_f16(
    const float* __restrict__ in, _Float16* __restrict__ out, int n) {
  int i = blockIdx.x * 256 + threadIdx.x;
  if (i < n) out[i] = (_Float16)in[i];
}

// ---------------------------------------------------------------------------
// WMMA projection: out = W @ x[b] + bias, f16 output, f32 accumulation.
// Block: 128 threads (4 waves), one 16-wide p tile per block.
// Per 32-deep C chunk, x[c:c+32, p:p+16] is transposed into LDS as f16.
//
// PMAJOR=true  (q,k; OTPW=1):  D[o,p] = W(A, global) x xT(B, LDS)
//                              -> out[b][p][o] rows, contiguous v8h stores.
// PMAJOR=false (v;   OTPW=8):  D[p,o] = xT(A, LDS) x W^T(B, global)
//                              -> out[b][o][p] rows, contiguous v8h stores;
//                              one shared A-frag, 8 o-tiles per wave.
// ---------------------------------------------------------------------------
template <int OTPW, bool PMAJOR>
__global__ __launch_bounds__(128) void pa_proj_wmma(
    const _Float16* __restrict__ Wf, const float* __restrict__ bias,
    const float* __restrict__ x, _Float16* __restrict__ out,
    int Cdim, int Pdim, int Odim) {
  __shared__ _Float16 xT[16][40];  // [p][c] f16, padded stride vs 64 banks

  const int b     = blockIdx.z;
  const int p0    = blockIdx.x * 16;
  const int tid   = threadIdx.x;
  const int lane  = tid & 31;
  const int wave  = tid >> 5;
  const int lo16  = lane & 15;
  const int hi    = lane >> 4;
  const int obase = wave * (OTPW * 16);

  const int r  = tid >> 2;       // c-row within chunk (0..31)
  const int pq = (tid & 3) * 4;  // p-column group (0,4,8,12)

  v8f acc[OTPW];
#pragma unroll
  for (int t = 0; t < OTPW; ++t) acc[t] = {};

  const float* xrow = x + ((size_t)b * Cdim + r) * Pdim + p0 + pq;

  for (int c0 = 0; c0 < Cdim; c0 += 32) {
    // stage x[c0+r][p0+pq..+3] -> xT[p][c] (f16, transposed)
    v4f xv = *(const v4f*)(xrow + (size_t)c0 * Pdim);
    __syncthreads();
#pragma unroll
    for (int i = 0; i < 4; ++i) xT[pq + i][r] = (_Float16)xv[i];
    __syncthreads();

    if constexpr (PMAJOR) {
      // B-frag: lane p = lo16, c halfs hi*8..+7 and 16+hi*8..+7 (two 16B LDS reads)
      v8h blo = *(const v8h*)(&xT[lo16][hi * 8]);
      v8h bhi = *(const v8h*)(&xT[lo16][16 + hi * 8]);
      v16h bf = cat8(blo, bhi);
      // A-frag: lane o = obase+lo16, c halfs hi*8..+7 and 16+hi*8..+7
      const _Float16* wr = Wf + (size_t)(obase + lo16) * Cdim + c0;
      v8h alo = *(const v8h*)(wr + hi * 8);
      v8h ahi = *(const v8h*)(wr + 16 + hi * 8);
      v16h af = cat8(alo, ahi);
      acc[0] = __builtin_amdgcn_wmma_f32_16x16x32_f16(
          false, af, false, bf, (short)0, acc[0], false, false);
    } else {
      // A-frag (shared by all o-tiles): lane p = lo16, c halfs hi*16..+15
      v8h alo = *(const v8h*)(&xT[lo16][hi * 16]);
      v8h ahi = *(const v8h*)(&xT[lo16][hi * 16 + 8]);
      v16h af = cat8(alo, ahi);
#pragma unroll
      for (int t = 0; t < OTPW; ++t) {
        // B-frag of W^T: lane o fixed, consecutive c -> contiguous W row 32B
        const _Float16* wr =
            Wf + (size_t)(obase + t * 16 + lo16) * Cdim + c0 + hi * 16;
        v16h bf = *(const v16h*)wr;
        acc[t] = __builtin_amdgcn_wmma_f32_16x16x32_f16(
            false, af, false, bf, (short)0, acc[t], false, false);
      }
    }
  }

  if constexpr (PMAJOR) {
    // D[o,p]: lane p = p0+lo16, o = obase + j + 8*hi -> contiguous o
    v8h o8;
#pragma unroll
    for (int j = 0; j < 8; ++j)
      o8[j] = (_Float16)(acc[0][j] + bias[obase + hi * 8 + j]);
    *(v8h*)(out + ((size_t)b * Pdim + p0 + lo16) * Odim + obase + hi * 8) = o8;
  } else {
    // D[p,o]: lane o = obase+t*16+lo16, p = p0 + j + 8*hi -> contiguous p
#pragma unroll
    for (int t = 0; t < OTPW; ++t) {
      const int o = obase + t * 16 + lo16;
      const float bo = bias[o];
      v8h o8;
#pragma unroll
      for (int j = 0; j < 8; ++j) o8[j] = (_Float16)(acc[t][j] + bo);
      *(v8h*)(out + ((size_t)b * Odim + o) * Pdim + p0 + hi * 8) = o8;
    }
  }
}

// ---------------------------------------------------------------------------
// Fused flash attention + output:  out = alpha * (softmax(qk) @ v^T)^T + x2
// Block: 128 threads (4 waves). One block per (b, 16-row p2 tile).
// Each wave owns c in [wave*128, wave*128+128) -> 8 f32 acc tiles.
// Energy computed transposed: ET[p1,p2] = KT(A) x Q^T(B); its D-layout is
// lane-for-lane the A-fragment layout of P for the PV WMMA.
// ---------------------------------------------------------------------------
__global__ __launch_bounds__(128) void pa_attention(
    const _Float16* __restrict__ qF,   // [B, P2, 64]
    const _Float16* __restrict__ ktF,  // [B, P1, 64]  (k transposed)
    const _Float16* __restrict__ vF,   // [B, C, P1]
    const float* __restrict__ x2, const float* __restrict__ alphaPtr,
    float* __restrict__ out) {
  const int b      = blockIdx.y;
  const int p2base = blockIdx.x * 16;
  const int lane   = threadIdx.x & 31;
  const int wave   = threadIdx.x >> 5;
  const int lo16   = lane & 15;
  const int hi     = lane >> 4;   // 0/1 half of wave
  const int cwbase = wave * 128;  // this wave's C slice

  // Q B-fragments (B = Q^T [cqk x p2]); lane: p2 = p2base+lo16,
  // kstep s: halfs cqk = s*32 + hi*16 .. +15 (contiguous 32B).
  const _Float16* qrow = qF + (size_t)(b * P2_ + p2base + lo16) * CQK_;
  const v16h qb0 = *(const v16h*)(qrow + hi * 16);
  const v16h qb1 = *(const v16h*)(qrow + 32 + hi * 16);

  v8f acc[8];
#pragma unroll
  for (int t = 0; t < 8; ++t) acc[t] = {};

  float m_run = -1e30f;
  float s_run = 0.f;

  const _Float16* ktB = ktF + (size_t)b * P1_ * CQK_;
  const _Float16* vB  = vF + (size_t)b * C_ * P1_;

  for (int it = 0; it < P1_ / 32; ++it) {
    const int p1base = it * 32;

    // --- ET tiles: [16 p1 x 16 p2], subtile 0 = p1 0..15, 1 = p1 16..31 ---
    const _Float16* k0 = ktB + (size_t)(p1base + lo16) * CQK_;
    const _Float16* k1 = ktB + (size_t)(p1base + 16 + lo16) * CQK_;
    // prefetch next chunk's K^T rows (global_prefetch_b8)
    if (it + 1 < P1_ / 32) {
      __builtin_prefetch(k0 + 32 * CQK_, 0, 1);
      __builtin_prefetch(k1 + 32 * CQK_, 0, 1);
    }
    v8f et0 = {}, et1 = {};
#pragma unroll
    for (int s = 0; s < 2; ++s) {
      const int ko = s * 32 + hi * 16;  // A-frag halfs K=ko..+15
      v8h a0lo = *(const v8h*)(k0 + ko);
      v8h a0hi = *(const v8h*)(k0 + ko + 8);
      v8h a1lo = *(const v8h*)(k1 + ko);
      v8h a1hi = *(const v8h*)(k1 + ko + 8);
      v16h a0 = cat8(a0lo, a0hi);
      v16h a1 = cat8(a1lo, a1hi);
      const v16h qb = s ? qb1 : qb0;
      et0 = __builtin_amdgcn_wmma_f32_16x16x32_f16(false, a0, false, qb,
                                                   (short)0, et0, false, false);
      et1 = __builtin_amdgcn_wmma_f32_16x16x32_f16(false, a1, false, qb,
                                                   (short)0, et1, false, false);
    }

    // --- online softmax: each lane holds fixed p2 = lo16, 16 p1 values ---
    float mc = et0[0];
#pragma unroll
    for (int i = 0; i < 8; ++i) {
      mc = fmaxf(mc, et0[i]);
      mc = fmaxf(mc, et1[i]);
    }
    mc = fmaxf(mc, __shfl_xor(mc, 16, 32));  // merge the two lane-halves
    const float m_new = fmaxf(m_run, mc);
    const float corr  = __expf(m_run - m_new);

    float rs = 0.f;
    v16h pa;  // A-fragment of P: halfs 0..7 = et0 rows, 8..15 = et1 rows
#pragma unroll
    for (int i = 0; i < 8; ++i) {
      float p0 = __expf(et0[i] - m_new);
      float p1 = __expf(et1[i] - m_new);
      rs += p0 + p1;
      pa[i]     = (_Float16)p0;
      pa[8 + i] = (_Float16)p1;
    }
    rs += __shfl_xor(rs, 16, 32);
    s_run = s_run * corr + rs;
    m_run = m_new;

    // --- rescale accumulators: acc row p2 = j + 8*hi -> broadcast corr ---
    float cj[8];
#pragma unroll
    for (int j = 0; j < 8; ++j) cj[j] = __shfl(corr, j + hi * 8, 32);
#pragma unroll
    for (int t = 0; t < 8; ++t)
#pragma unroll
      for (int j = 0; j < 8; ++j) acc[t][j] *= cj[j];

    // --- acc += P x V : B-frag of V is two 16B row reads, p1 hi*8..+7 and 16+hi*8..+7 ---
#pragma unroll
    for (int t = 0; t < 8; ++t) {
      const _Float16* vrow =
          vB + (size_t)(cwbase + t * 16 + lo16) * P1_ + p1base + hi * 8;
      v16h vb = cat8(*(const v8h*)vrow, *(const v8h*)(vrow + 16));
      acc[t] = __builtin_amdgcn_wmma_f32_16x16x32_f16(
          false, pa, false, vb, (short)0, acc[t], false, false);
    }
  }

  // --- epilogue: out[b,c,p2] = alpha * acc/s + x2[b,c,p2] ---
  const float alpha = alphaPtr[0];
  const float inv   = 1.f / s_run;
  float ij[8];
#pragma unroll
  for (int j = 0; j < 8; ++j) ij[j] = __shfl(inv, j + hi * 8, 32);

#pragma unroll
  for (int t = 0; t < 8; ++t) {
    const int c       = cwbase + t * 16 + lo16;
    const size_t base = (size_t)(b * C_ + c) * P2_ + p2base + hi * 8;
    v4f xa = *(const v4f*)(x2 + base);
    v4f xb = *(const v4f*)(x2 + base + 4);
    v4f o0, o1;
#pragma unroll
    for (int j = 0; j < 4; ++j) {
      o0[j] = alpha * acc[t][j] * ij[j] + xa[j];
      o1[j] = alpha * acc[t][4 + j] * ij[4 + j] + xb[j];
    }
    *(v4f*)(out + base)     = o0;
    *(v4f*)(out + base + 4) = o1;
  }
}

// ---------------------------------------------------------------------------
extern "C" void kernel_launch(void* const* d_in, const int* in_sizes, int n_in,
                              void* d_out, int out_size, void* d_ws,
                              size_t ws_size, hipStream_t stream) {
  const float* x1    = (const float*)d_in[0];
  const float* x2    = (const float*)d_in[1];
  const float* Wq    = (const float*)d_in[2];
  const float* bq    = (const float*)d_in[3];
  const float* Wk    = (const float*)d_in[4];
  const float* bk    = (const float*)d_in[5];
  const float* Wv    = (const float*)d_in[6];
  const float* bv    = (const float*)d_in[7];
  const float* alpha = (const float*)d_in[8];

  _Float16* qws  = (_Float16*)d_ws;                  // [B,P2,64]   8 MB
  _Float16* ktws = qws + (size_t)B_ * P2_ * CQK_;    // [B,P1,64]   4 MB
  _Float16* vws  = ktws + (size_t)B_ * P1_ * CQK_;   // [B,C,P1]   32 MB
  _Float16* wqf  = vws + (size_t)B_ * C_ * P1_;      // [64,512]
  _Float16* wkf  = wqf + (size_t)CQK_ * C_;          // [64,512]
  _Float16* wvf  = wkf + (size_t)CQK_ * C_;          // [512,512]

  const int nqk = CQK_ * C_;  // 32768
  const int nv  = C_ * C_;    // 262144
  pa_cvt_f16<<<(nqk + 255) / 256, 256, 0, stream>>>(Wq, wqf, nqk);
  pa_cvt_f16<<<(nqk + 255) / 256, 256, 0, stream>>>(Wk, wkf, nqk);
  pa_cvt_f16<<<(nv + 255) / 256, 256, 0, stream>>>(Wv, wvf, nv);

  pa_proj_wmma<1, true><<<dim3(P2_ / 16, 1, B_), 128, 0, stream>>>(
      wqf, bq, x2, qws, C_, P2_, CQK_);
  pa_proj_wmma<1, true><<<dim3(P1_ / 16, 1, B_), 128, 0, stream>>>(
      wkf, bk, x1, ktws, C_, P1_, CQK_);
  pa_proj_wmma<8, false><<<dim3(P1_ / 16, 1, B_), 128, 0, stream>>>(
      wvf, bv, x1, vws, C_, P1_, C_);

  pa_attention<<<dim3(P2_ / 16, B_), 128, 0, stream>>>(qws, ktws, vws, x2,
                                                       alpha, (float*)d_out);
}
